// Split_KL_1271310319848
// MI455X (gfx1250) — matrix operand order repaired and checked by reading, not compiled, not run on hardware
//
#include <hip/hip_runtime.h>
#include <hip/hip_bf16.h>

typedef _Float16 v16h __attribute__((ext_vector_type(16)));
typedef _Float16 v8h  __attribute__((ext_vector_type(8)));
typedef float    v8f  __attribute__((ext_vector_type(8)));

#define BB   512     // batch
#define DD   256     // feature dim
#define PP   196     // spatial positions (14*14)
#define CAA  8       // attention channels
#define NROW 1024    // 2*B
#define INV_TEMP 2.0f

// ---------------------------------------------------------------- zero output
__global__ void zero_kernel(float* out, int n) {
    int i = blockIdx.x * blockDim.x + threadIdx.x;
    if (i < n) out[i] = 0.0f;
}

// ------------------------------------------------- weight[p] = mean(att)/P
__global__ void att_weight_kernel(const float* __restrict__ att,
                                  float* __restrict__ weight) {
    const int p = blockIdx.x;                 // 0..195
    float s = 0.0f;
    for (int idx = threadIdx.x; idx < BB * CAA; idx += blockDim.x)
        s += att[(size_t)idx * PP + p];
    __shared__ float red[8];
    for (int off = 16; off; off >>= 1) s += __shfl_down(s, off, 32);
    if ((threadIdx.x & 31) == 0) red[threadIdx.x >> 5] = s;
    __syncthreads();
    if (threadIdx.x == 0) {
        float t = 0.0f;
        for (int w = 0; w < 8; w++) t += red[w];
        weight[p] = t / ((float)BB * (float)PP);   // mean over B, then /P
    }
}

// ------------------------ local feature rows: softmax -> L2 norm -> f16 Z rows
__global__ void local_norm_kernel(const float* __restrict__ outputs,
                                  _Float16* __restrict__ Z) {
    const int p = blockIdx.x;        // 0..195
    const int b = blockIdx.y;        // 0..511
    const int d = threadIdx.x;       // 0..255
    __shared__ float red[8];
    const int wave = d >> 5, lane = d & 31;

    float v = outputs[((size_t)b * DD + d) * PP + p];

    // ---- max
    float m = v;
    for (int off = 16; off; off >>= 1) m = fmaxf(m, __shfl_xor(m, off, 32));
    if (lane == 0) red[wave] = m;
    __syncthreads();
    m = red[0];
    for (int w = 1; w < 8; w++) m = fmaxf(m, red[w]);
    __syncthreads();

    // ---- exp-sum
    float e = __expf(v - m);
    float s = e;
    for (int off = 16; off; off >>= 1) s += __shfl_xor(s, off, 32);
    if (lane == 0) red[wave] = s;
    __syncthreads();
    s = 0.0f;
    for (int w = 0; w < 8; w++) s += red[w];
    __syncthreads();

    float sm = e / s;                 // softmaxed entry

    // ---- L2 norm
    float q = sm * sm;
    for (int off = 16; off; off >>= 1) q += __shfl_xor(q, off, 32);
    if (lane == 0) red[wave] = q;
    __syncthreads();
    q = 0.0f;
    for (int w = 0; w < 8; w++) q += red[w];

    float zn = sm / (__fsqrt_rn(q) + 1e-8f);
    Z[((size_t)p * NROW + b) * DD + d] = (_Float16)zn;
}

// ----- global feature: carry g through 196 repeated softmaxes, emit normalized
__global__ void global_seq_kernel(const float* __restrict__ g0,
                                  _Float16* __restrict__ Z) {
    const int b = blockIdx.x;        // 0..511
    const int d = threadIdx.x;       // 0..255
    __shared__ float red[8];
    const int wave = d >> 5, lane = d & 31;

    float g = g0[(size_t)b * DD + d];

    for (int p = 0; p < PP; p++) {
        // ---- softmax (carry)
        float m = g;
        for (int off = 16; off; off >>= 1) m = fmaxf(m, __shfl_xor(m, off, 32));
        if (lane == 0) red[wave] = m;
        __syncthreads();
        m = red[0];
        for (int w = 1; w < 8; w++) m = fmaxf(m, red[w]);
        __syncthreads();

        float e = __expf(g - m);
        float s = e;
        for (int off = 16; off; off >>= 1) s += __shfl_xor(s, off, 32);
        if (lane == 0) red[wave] = s;
        __syncthreads();
        s = 0.0f;
        for (int w = 0; w < 8; w++) s += red[w];
        __syncthreads();

        g = e / s;                    // carry forward (un-normalized softmax)

        // ---- L2-normalized copy into Z
        float q = g * g;
        for (int off = 16; off; off >>= 1) q += __shfl_xor(q, off, 32);
        if (lane == 0) red[wave] = q;
        __syncthreads();
        q = 0.0f;
        for (int w = 0; w < 8; w++) q += red[w];
        __syncthreads();

        float zn = g / (__fsqrt_rn(q) + 1e-8f);
        Z[((size_t)p * NROW + BB + b) * DD + d] = (_Float16)zn;
    }
}

// ---------------------------------------------------- main WMMA NT-Xent kernel
// grid: (32 row-stripes of 32 rows, 196 steps), block: 128 threads = 4 waves.
// Each wave: 2 resident A tiles (32 rows x K=256), 16 column tiles (256 cols),
// ping-pong B fragment sets (static indexing only!), 2 interleaved WMMA chains.
__global__ __launch_bounds__(128) void ntxent_kernel(
        const _Float16* __restrict__ Z,
        const float* __restrict__ weight,
        float* __restrict__ out) {
    const int p      = blockIdx.y;
    const int stripe = blockIdx.x;            // 32-row stripe, 0..31
    const int wave   = threadIdx.x >> 5;      // 0..3
    const int lane   = threadIdx.x & 31;
    const int hs     = lane >> 4;             // half-wave select
    const int l15    = lane & 15;

    const _Float16* Zp = Z + (size_t)p * NROW * DD;
    const int rowBase = stripe * 32;

    // ---- resident A fragments for rows rowBase..+15 (A0) and +16..+31 (A1)
    // 16-bit A 16x32 layout: lane(l15)=row M; elems 0..7 -> K=hs*8.., 8..15 -> K=hs*8+16..
    v16h A0[8], A1[8];
    {
        const _Float16* a0 = Zp + (size_t)(rowBase + l15) * DD;
        const _Float16* a1 = a0 + (size_t)16 * DD;
#pragma unroll
        for (int f = 0; f < 8; f++) {
            const int base = f * 32 + hs * 8;
            v8h lo0 = *(const v8h*)(a0 + base);
            v8h hi0 = *(const v8h*)(a0 + base + 16);
            v8h lo1 = *(const v8h*)(a1 + base);
            v8h hi1 = *(const v8h*)(a1 + base + 16);
#pragma unroll
            for (int e = 0; e < 8; e++) {
                A0[f][e] = lo0[e]; A0[f][8 + e] = hi0[e];
                A1[f][e] = lo1[e]; A1[f][8 + e] = hi1[e];
            }
        }
    }

    float rowAcc[16];
#pragma unroll
    for (int k = 0; k < 16; k++) rowAcc[k] = 0.0f;
    float posAcc = 0.0f;

    const int ctBeg = wave * 16, ctEnd = ctBeg + 16;
    const int kOff = hs * 8;                  // B table-order K chunk select
    const int grow0 = rowBase + hs * 8;       // C0 rows = grow0 + k
    const int grow1 = grow0 + 16;             // C1 rows = grow1 + k

    // ---- B fragment loader: static destination, no dynamic array indexing
    auto loadB = [&](v16h (&Bf)[8], int ctv) {
        const _Float16* bptr = Zp + (size_t)(ctv * 16 + l15) * DD;
#pragma unroll
        for (int f = 0; f < 8; f++) {
            v8h lo = *(const v8h*)(bptr + f * 32 + kOff);
            v8h hi = *(const v8h*)(bptr + f * 32 + kOff + 16);
            Bf[f] = __builtin_shufflevector(lo, hi, 0,1,2,3,4,5,6,7,8,9,10,11,12,13,14,15);
        }
    };

    // ---- one 32x16 tile: two interleaved WMMA chains + exp/pos epilogue
    auto tile = [&](const v16h (&Bf)[8], int ctv) {
        v8f C0 = {}, C1 = {};
#pragma unroll
        for (int f = 0; f < 8; f++) {
            C0 = __builtin_amdgcn_wmma_f32_16x16x32_f16(
                    false, A0[f], false, Bf[f], (short)0, C0, false, false);
            C1 = __builtin_amdgcn_wmma_f32_16x16x32_f16(
                    false, A1[f], false, Bf[f], (short)0, C1, false, false);
        }
        const int gcol = ctv * 16 + l15;
#pragma unroll
        for (int k = 0; k < 8; k++) {
            const int r0 = grow0 + k;
            const float s0 = C0[k] * INV_TEMP;           // sim in [0,2]
            if (gcol != r0) rowAcc[k] += __expf(s0);     // diagonal masked
            if (gcol == (r0 ^ BB)) posAcc += s0;         // positive pair

            const int r1 = grow1 + k;
            const float s1 = C1[k] * INV_TEMP;
            if (gcol != r1) rowAcc[8 + k] += __expf(s1);
            if (gcol == (r1 ^ BB)) posAcc += s1;
        }
    };

    // ---- ping-pong pipeline over 16 column tiles (8 iterations x 2 tiles)
    v16h Ba[8], Bb[8];
    loadB(Ba, ctBeg);
#pragma unroll 1
    for (int ct = ctBeg; ct < ctEnd; ct += 2) {
        loadB(Bb, ct + 1);            // prefetch while Ba chains execute
        tile(Ba, ct);
        if (ct + 2 < ctEnd) loadB(Ba, ct + 2);  // prefetch while Bb chains run
        tile(Bb, ct + 1);
    }

    // ---- reduce row sums across the 16 lanes of each half-wave
#pragma unroll
    for (int k = 0; k < 16; k++) {
        float v = rowAcc[k];
        for (int off = 1; off < 16; off <<= 1) v += __shfl_xor(v, off, 32);
        rowAcc[k] = v;
    }
    for (int off = 1; off < 32; off <<= 1) posAcc += __shfl_xor(posAcc, off, 32);

    __shared__ float ldsRow[32];
    __shared__ float ldsPos;
    if (threadIdx.x < 32) ldsRow[threadIdx.x] = 0.0f;
    if (threadIdx.x == 0) ldsPos = 0.0f;
    __syncthreads();

    if (l15 == 0) {                           // lanes 0 and 16 of each wave
#pragma unroll
        for (int k = 0; k < 8; k++) {
            atomicAdd(&ldsRow[hs * 8 + k],      rowAcc[k]);       // rows 0..15
            atomicAdd(&ldsRow[16 + hs * 8 + k], rowAcc[8 + k]);   // rows 16..31
        }
        if (hs == 0) atomicAdd(&ldsPos, posAcc);
    }
    __syncthreads();

    if (threadIdx.x == 0) {
        float acc = 0.0f;
#pragma unroll
        for (int t = 0; t < 32; t++) acc += __logf(ldsRow[t]);
        acc -= ldsPos;                        // sum(logS - pos) over 32 rows
        const float lp = acc * weight[p] * (1.0f / (float)NROW);
        atomicAdd(out, lp);
    }
}

// ----------------------------------------------------------------------------
extern "C" void kernel_launch(void* const* d_in, const int* in_sizes, int n_in,
                              void* d_out, int out_size, void* d_ws, size_t ws_size,
                              hipStream_t stream) {
    const float* gfeat = (const float*)d_in[0];   // [512, 256]
    const float* att   = (const float*)d_in[1];   // [512, 8, 14, 14]
    const float* outs  = (const float*)d_in[2];   // [512, 256, 14, 14]
    float* out = (float*)d_out;

    char* ws = (char*)d_ws;
    float*    weight = (float*)ws;                       // 196 floats
    _Float16* Z      = (_Float16*)(ws + 1024);           // [196][1024][256] f16

    zero_kernel<<<1, 64, 0, stream>>>(out, out_size);
    att_weight_kernel<<<PP, 256, 0, stream>>>(att, weight);
    local_norm_kernel<<<dim3(PP, BB), DD, 0, stream>>>(outs, Z);
    global_seq_kernel<<<BB, DD, 0, stream>>>(gfeat, Z);
    ntxent_kernel<<<dim3(NROW / 32, PP), 128, 0, stream>>>(Z, weight, out);
}
